// BertSelfAttention_46299747451079
// MI455X (gfx1250) — matrix operand order repaired and checked by reading, not compiled, not run on hardware
//
#include <hip/hip_runtime.h>

typedef __bf16 bf16;
typedef __attribute__((ext_vector_type(16))) __bf16 v16bf;
typedef __attribute__((ext_vector_type(8)))  __bf16 v8bf;
typedef __attribute__((ext_vector_type(4)))  __bf16 v4bf;
typedef __attribute__((ext_vector_type(8)))  float  v8f;
typedef __attribute__((ext_vector_type(4)))  float  v4f;

#define B_   4
#define S_   2048
#define HID_ 1024
#define H_   16
#define D_   64
#define LOG2E 1.44269504088896340736f

union FragT { v16bf v; v8bf h[2]; };

// ---- gfx1250 async global->LDS copy (ASYNCcnt) --------------------
__device__ __forceinline__ void async_ld_b128(const bf16* g, const bf16* l) {
  asm volatile("global_load_async_to_lds_b128 %0, %1, off"
               :: "v"((unsigned)(size_t)(l)),
                  "v"((unsigned long long)(size_t)(g))
               : "memory");
}
#define WAIT_ASYNC(N) asm volatile("s_wait_asynccnt %0" :: "i"(N) : "memory")

__device__ __forceinline__ bf16 f2bf(float f) {
  unsigned u = __builtin_bit_cast(unsigned, f);
  unsigned r = u + 0x7fffu + ((u >> 16) & 1u);
  unsigned short hh = (unsigned short)(r >> 16);
  return __builtin_bit_cast(__bf16, hh);
}

#define WMMA_BF16(A, Bv, C) \
  __builtin_amdgcn_wmma_f32_16x16x32_bf16(false, (A), false, (Bv), (short)0, (C), false, false)

// ---------------- Kernel 0: fp32 -> bf16 pack (RNE) ----------------
__global__ __launch_bounds__(256) void cvt_f32_bf16(const float* __restrict__ src,
                                                    bf16* __restrict__ dst, int n) {
  int i = (blockIdx.x * 256 + threadIdx.x) * 4;
  if (i + 3 < n) {
    v4f x = *(const v4f*)(src + i);
    v4bf o;
    o.x = f2bf(x.x); o.y = f2bf(x.y); o.z = f2bf(x.z); o.w = f2bf(x.w);
    *(v4bf*)(dst + i) = o;
  }
}

// ---------------- Kernel 1: fused QKV projection GEMM --------------
// Block: 8 waves x (32M x 64N) = 256M x 64N tile. W tile staged in LDS
// (async, double buffered). Per 32-k chunk: 4 global + 8 DS loads feed
// 8 independent back-to-back WMMAs.
__global__ __launch_bounds__(256, 1) void qkv_gemm(
    const bf16* __restrict__ xb,
    const bf16* __restrict__ wq, const bf16* __restrict__ wk, const bf16* __restrict__ wv,
    const float* __restrict__ bq, const float* __restrict__ bk, const float* __restrict__ bv,
    bf16* __restrict__ Qb, bf16* __restrict__ Kb, bf16* __restrict__ VTb) {
  __shared__ alignas(16) bf16 Wt[2][64 * 32];      // 4KB x2, 64 N-rows x 32 K

  const int lane = threadIdx.x & 31;
  const int wave = threadIdx.x >> 5;
  const int bid  = blockIdx.x;                     // 3 * 16 * 32 blocks

  const int mat = bid / (16 * 32);                 // 0:Q 1:K 2:V
  const int rem = bid % (16 * 32);
  const int nt  = rem / 32;                        // 64-wide N tile
  const int mtb = rem % 32;                        // 256-tall M block

  const bf16*  W    = (mat == 0) ? wq : (mat == 1) ? wk : wv;
  const float* bias = (mat == 0) ? bq : (mat == 1) ? bk : bv;

  const int row = lane & 15;
  const int hi  = lane >> 4;
  const int m0  = mtb * 256 + wave * 32;           // this wave: rows m0..m0+31
  const int n0  = nt * 64;

  // staging map: thread -> (W row, 16B chunk); 256 thr * 16B = 4KB/chunk
  const int trow = threadIdx.x >> 2;               // 0..63
  const int tchk = threadIdx.x & 3;                // 0..3
  const bf16* wsrc = W + (size_t)(n0 + trow) * HID_ + tchk * 8;
  bf16* wdst0 = &Wt[0][trow * 32 + tchk * 8];
  bf16* wdst1 = &Wt[1][trow * 32 + tchk * 8];

  v8f acc0[4] = {}, acc1[4] = {};
  const bf16* arow0 = xb + (size_t)(m0 + row) * HID_;
  const bf16* arow1 = xb + (size_t)(m0 + 16 + row) * HID_;

  async_ld_b128(wsrc, wdst0);                      // prologue stage k=0
  int cur = 0;
  for (int k = 0; k < HID_; k += 32) {
    if (k + 32 < HID_) {
      async_ld_b128(wsrc + (k + 32), cur ? wdst0 : wdst1);
      WAIT_ASYNC(1);                               // cur buffer landed
    } else {
      WAIT_ASYNC(0);
    }
    __syncthreads();

    FragT A0, A1;
    A0.h[0] = *(const v8bf*)(arow0 + k + hi * 8);
    A0.h[1] = *(const v8bf*)(arow0 + k + 16 + hi * 8);
    A1.h[0] = *(const v8bf*)(arow1 + k + hi * 8);
    A1.h[1] = *(const v8bf*)(arow1 + k + 16 + hi * 8);

    const bf16* wl = Wt[cur];
    FragT Bf[4];
#pragma unroll
    for (int t = 0; t < 4; ++t) {                  // batch all DS loads first
      const bf16* p = wl + (t * 16 + row) * 32 + hi * 8;
      Bf[t].h[0] = *(const v8bf*)p;
      Bf[t].h[1] = *(const v8bf*)(p + 16);
    }
#pragma unroll
    for (int t = 0; t < 4; ++t) {                  // 8 independent WMMAs
      acc0[t] = WMMA_BF16(A0.v, Bf[t].v, acc0[t]);
      acc1[t] = WMMA_BF16(A1.v, Bf[t].v, acc1[t]);
    }
    __syncthreads();                               // done reading cur
    cur ^= 1;
  }

#pragma unroll
  for (int t = 0; t < 4; ++t) {
    const int nglob = n0 + t * 16 + row;
    const float bval = bias[nglob];
    const int hd = nglob >> 6, d = nglob & 63;
#pragma unroll
    for (int i = 0; i < 2; ++i) {
      const v8f* accp = i ? acc1 : acc0;
#pragma unroll
      for (int r = 0; r < 8; ++r) {
        const int mglob = m0 + i * 16 + r + hi * 8;
        const int b = mglob >> 11, s = mglob & 2047;
        const bf16 o = f2bf(accp[t][r] + bval);
        if (mat == 2)
          VTb[((size_t)(b * H_ + hd) * D_ + d) * S_ + s] = o;
        else if (mat == 0)
          Qb[((size_t)(b * H_ + hd) * S_ + s) * D_ + d] = o;
        else
          Kb[((size_t)(b * H_ + hd) * S_ + s) * D_ + d] = o;
      }
    }
  }
}

// ---------------- Kernel 2: flash attention (online softmax) -------
// All 8 waves of a block share one (b,h): K/V chunks staged once per
// block in LDS (async, double buffered); per-wave 16-row q tiles.
__global__ __launch_bounds__(256, 1) void attention(
    const bf16* __restrict__ Qb, const bf16* __restrict__ Kb, const bf16* __restrict__ VTb,
    const float* __restrict__ mask, float* __restrict__ out) {
  __shared__ alignas(16) bf16 Kt[2][32 * 64];      // 32 keys x 64 d
  __shared__ alignas(16) bf16 Vt[2][64 * 32];      // 64 d x 32 keys (V^T chunk)
  __shared__ alignas(16) bf16 Pb[8][16 * 32];      // per-wave P bounce

  const int lane = threadIdx.x & 31;
  const int wave = threadIdx.x >> 5;
  const int wid  = blockIdx.x * 8 + wave;          // 8192 waves; block uniform in bh
  const int qt = wid & 127;
  const int bh = wid >> 7;
  const int b = bh >> 4, hh = bh & 15;

  const int row = lane & 15;
  const int hi  = lane >> 4;

  const bf16* Qhead = Qb  + (size_t)(b * H_ + hh) * S_ * D_;
  const bf16* Khead = Kb  + (size_t)(b * H_ + hh) * S_ * D_;
  const bf16* Vhead = VTb + (size_t)(b * H_ + hh) * D_ * S_;
  const float* mrow = mask + (size_t)b * S_;

  // staging maps (256 threads, 16B each): K chunk 4KB, V chunk 4KB
  const int tkr = threadIdx.x >> 3, tkc = threadIdx.x & 7;   // K: 32 rows x 128B
  const int tvr = threadIdx.x >> 2, tvc = threadIdx.x & 3;   // V: 64 rows x 64B
  const bf16* ksrc = Khead + (size_t)tkr * D_ + tkc * 8;
  const bf16* vsrc = Vhead + (size_t)tvr * S_ + tvc * 8;
  bf16* kdst[2] = { &Kt[0][tkr * 64 + tkc * 8], &Kt[1][tkr * 64 + tkc * 8] };
  bf16* vdst[2] = { &Vt[0][tvr * 32 + tvc * 8], &Vt[1][tvr * 32 + tvc * 8] };

  FragT qf[2];
  {
    const bf16* qr = Qhead + (size_t)(qt * 16 + row) * D_;
#pragma unroll
    for (int c = 0; c < 2; ++c) {
      qf[c].h[0] = *(const v8bf*)(qr + c * 32 + hi * 8);
      qf[c].h[1] = *(const v8bf*)(qr + c * 32 + 16 + hi * 8);
    }
  }

  float mstate[8], lstate[8];
#pragma unroll
  for (int r = 0; r < 8; ++r) { mstate[r] = -3.0e38f; lstate[r] = 0.f; }
  v8f o[4] = {};

  const float kS = 0.125f * LOG2E;
  bf16* P = Pb[wave];

  async_ld_b128(ksrc, kdst[0]);                    // prologue stage j0=0
  async_ld_b128(vsrc, vdst[0]);
  int cur = 0;

  for (int j0 = 0; j0 < S_; j0 += 32) {
    if (j0 + 32 < S_) {
      async_ld_b128(ksrc + (size_t)(j0 + 32) * D_, kdst[cur ^ 1]);
      async_ld_b128(vsrc + j0 + 32, vdst[cur ^ 1]);
      WAIT_ASYNC(2);                               // cur K+V landed
    } else {
      WAIT_ASYNC(0);
    }
    __syncthreads();

    const bf16* Kc = Kt[cur];
    const bf16* Vc = Vt[cur];

    // ---- scores: batch all 4 K fragments, then 4 WMMAs
    FragT kf[2][2];
#pragma unroll
    for (int a = 0; a < 2; ++a)
#pragma unroll
      for (int c = 0; c < 2; ++c) {
        const bf16* kr = Kc + (a * 16 + row) * 64 + c * 32 + hi * 8;
        kf[a][c].h[0] = *(const v8bf*)(kr);
        kf[a][c].h[1] = *(const v8bf*)(kr + 16);
      }
    v8f s0 = {}, s1 = {};
    s0 = WMMA_BF16(qf[0].v, kf[0][0].v, s0);
    s1 = WMMA_BF16(qf[0].v, kf[1][0].v, s1);
    s0 = WMMA_BF16(qf[1].v, kf[0][1].v, s0);
    s1 = WMMA_BF16(qf[1].v, kf[1][1].v, s1);

    const float mk0 = mrow[j0 + row] * LOG2E;
    const float mk1 = mrow[j0 + 16 + row] * LOG2E;

    // ---- online softmax (log2 domain)
    float sc0[8], sc1[8], mx[8];
#pragma unroll
    for (int r = 0; r < 8; ++r) {
      sc0[r] = s0[r] * kS + mk0;
      sc1[r] = s1[r] * kS + mk1;
      mx[r] = fmaxf(sc0[r], sc1[r]);
    }
#pragma unroll
    for (int off = 1; off < 16; off <<= 1) {
#pragma unroll
      for (int r = 0; r < 8; ++r) mx[r] = fmaxf(mx[r], __shfl_xor(mx[r], off, 16));
    }
    float alpha[8], rs[8];
#pragma unroll
    for (int r = 0; r < 8; ++r) {
      const float mn = fmaxf(mstate[r], mx[r]);
      alpha[r] = exp2f(mstate[r] - mn);
      mstate[r] = mn;
      sc0[r] = exp2f(sc0[r] - mn);
      sc1[r] = exp2f(sc1[r] - mn);
      rs[r] = sc0[r] + sc1[r];
    }
#pragma unroll
    for (int off = 1; off < 16; off <<= 1) {
#pragma unroll
      for (int r = 0; r < 8; ++r) rs[r] += __shfl_xor(rs[r], off, 16);
    }
#pragma unroll
    for (int r = 0; r < 8; ++r) {
      lstate[r] = lstate[r] * alpha[r] + rs[r];
#pragma unroll
      for (int t = 0; t < 4; ++t) o[t][r] *= alpha[r];
    }

    // ---- P: C-layout -> row-major LDS -> A-fragment (per-wave, DS in-order)
#pragma unroll
    for (int r = 0; r < 8; ++r) {
      const int m = r + hi * 8;
      P[m * 32 + row]      = f2bf(sc0[r]);
      P[m * 32 + 16 + row] = f2bf(sc1[r]);
    }
    asm volatile("s_wait_dscnt 0" ::: "memory");
    FragT pf;
    pf.h[0] = *(const v8bf*)(P + row * 32 + hi * 8);
    pf.h[1] = *(const v8bf*)(P + row * 32 + 16 + hi * 8);

    // ---- O += P V: batch all 4 V fragments, then 4 independent WMMAs
    FragT vf[4];
#pragma unroll
    for (int t = 0; t < 4; ++t) {
      const bf16* vr = Vc + (t * 16 + row) * 32 + hi * 8;
      vf[t].h[0] = *(const v8bf*)(vr);
      vf[t].h[1] = *(const v8bf*)(vr + 16);
    }
#pragma unroll
    for (int t = 0; t < 4; ++t)
      o[t] = WMMA_BF16(pf.v, vf[t].v, o[t]);

    __syncthreads();                               // done reading cur buffers
    cur ^= 1;
  }

  // ---- epilogue: normalize, scatter to [B,S,HID]
#pragma unroll
  for (int r = 0; r < 8; ++r) {
    const float inv = 1.0f / lstate[r];
    const int s = qt * 16 + r + hi * 8;
    float* orow = out + ((size_t)b * S_ + s) * HID_ + hh * D_;
#pragma unroll
    for (int t = 0; t < 4; ++t) orow[t * 16 + row] = o[t][r] * inv;
  }
}

// ---------------- Host launcher ------------------------------------
extern "C" void kernel_launch(void* const* d_in, const int* in_sizes, int n_in,
                              void* d_out, int out_size, void* d_ws, size_t ws_size,
                              hipStream_t stream) {
  const float* x    = (const float*)d_in[0];
  const float* mask = (const float*)d_in[1];
  const float* Wq   = (const float*)d_in[2];
  const float* bq   = (const float*)d_in[3];
  const float* Wk   = (const float*)d_in[4];
  const float* bk   = (const float*)d_in[5];
  const float* Wv   = (const float*)d_in[6];
  const float* bv   = (const float*)d_in[7];
  float* out = (float*)d_out;

  char* ws = (char*)d_ws;
  const size_t xN = (size_t)B_ * S_ * HID_;
  const size_t wN = (size_t)HID_ * HID_;
  const size_t qkvN = (size_t)B_ * H_ * S_ * D_;

  bf16* xb  = (bf16*)ws;                     ws += xN * 2;
  bf16* wqb = (bf16*)ws;                     ws += wN * 2;
  bf16* wkb = (bf16*)ws;                     ws += wN * 2;
  bf16* wvb = (bf16*)ws;                     ws += wN * 2;
  bf16* Qb  = (bf16*)ws;                     ws += qkvN * 2;
  bf16* Kb  = (bf16*)ws;                     ws += qkvN * 2;
  bf16* VTb = (bf16*)ws;                     ws += qkvN * 2;

  cvt_f32_bf16<<<(int)(xN / 4 / 256), 256, 0, stream>>>(x,  xb,  (int)xN);
  cvt_f32_bf16<<<(int)(wN / 4 / 256), 256, 0, stream>>>(Wq, wqb, (int)wN);
  cvt_f32_bf16<<<(int)(wN / 4 / 256), 256, 0, stream>>>(Wk, wkb, (int)wN);
  cvt_f32_bf16<<<(int)(wN / 4 / 256), 256, 0, stream>>>(Wv, wvb, (int)wN);

  // 3 mats * 16 n-tiles * 32 m-blocks(256 rows) = 1536 blocks
  qkv_gemm<<<1536, 256, 0, stream>>>(xb, wqb, wkb, wvb, bq, bk, bv, Qb, Kb, VTb);

  // B*H*(S/16) = 8192 waves / 8 per block
  attention<<<1024, 256, 0, stream>>>(Qb, Kb, VTb, mask, out);
}
